// AttnBlock_48473000903020
// MI455X (gfx1250) — matrix operand order repaired and checked by reading, not compiled, not run on hardware
//
#include <hip/hip_runtime.h>
#include <hip/hip_bf16.h>

typedef __attribute__((ext_vector_type(16))) _Float16 v16h;
typedef __attribute__((ext_vector_type(8)))  _Float16 v8h;
typedef __attribute__((ext_vector_type(2)))  _Float16 v2h;
typedef __attribute__((ext_vector_type(8)))  float    v8f;

#define BATCH 2
#define CCH   512
#define NHW   4096
#define NGRP  32
#define CGRP  16
#define JT    32

// ---------------------------------------------------------------------------
// WMMA fragment helpers, in this file's contiguous k order (wave32):
// A 16x32: M=lane%16, K(e)=e+16*(lane>=16)
//   rowp points at (row, kbase + 16*hi); 16 contiguous f16
// B 32x16: N=lane%16, K(e)=e+16*(lane>=16) -> contiguous 16 f16
// D 16x16: M=r+8*(lane>=16), N=lane%16
// ---------------------------------------------------------------------------
__device__ inline v16h load_a_frag(const _Float16* rowp) {
    v8h lo = *(const v8h*)rowp;
    v8h hi = *(const v8h*)(rowp + 8);
    v16h a;
#pragma unroll
    for (int e = 0; e < 8; ++e) { a[e] = lo[e]; a[e + 8] = hi[e]; }
    return a;
}

__device__ inline v8f wmma_f16(v16h a, v16h b, v8f c) {
    return __builtin_amdgcn_wmma_f32_16x16x32_f16(false, a, false, b, (short)0, c,
                                                  false, false);
}

// ---------------------------------------------------------------------------
// K0: f32 -> f16 weight conversion
// ---------------------------------------------------------------------------
__global__ void cvt_f16_kernel(const float* __restrict__ w, _Float16* __restrict__ o,
                               int n) {
    int i = blockIdx.x * blockDim.x + threadIdx.x;
    if (i < n) o[i] = (_Float16)w[i];
}

// ---------------------------------------------------------------------------
// K1: GroupNorm -> f16.  One block per (b, group): reduce 16*4096 elements.
// ---------------------------------------------------------------------------
__global__ __launch_bounds__(256) void groupnorm_kernel(
    const float* __restrict__ x, const float* __restrict__ gamma,
    const float* __restrict__ beta, _Float16* __restrict__ hn) {
    int bg  = blockIdx.x;
    int b   = bg / NGRP;
    int grp = bg % NGRP;
    const size_t base = ((size_t)b * CCH + (size_t)grp * CGRP) * NHW;
    const float* xp = x + base;
    const int NE = CGRP * NHW;  // 65536
    int tid = threadIdx.x;

    float s = 0.f, ss = 0.f;
    for (int i = tid; i < NE; i += 256) {
        float v = xp[i];
        s += v; ss += v * v;
    }
    __shared__ float r0[256], r1[256];
    r0[tid] = s; r1[tid] = ss;
    __syncthreads();
    for (int o = 128; o > 0; o >>= 1) {
        if (tid < o) { r0[tid] += r0[tid + o]; r1[tid] += r1[tid + o]; }
        __syncthreads();
    }
    float mean = r0[0] * (1.0f / NE);
    float var  = r1[0] * (1.0f / NE) - mean * mean;
    float rstd = rsqrtf(var + 1e-6f);

    _Float16* op = hn + base;
    for (int i = tid; i < NE; i += 256) {
        int c = grp * CGRP + (i >> 12);  // i / NHW
        float v = (xp[i] - mean) * rstd * gamma[c] + beta[c];
        op[i] = (_Float16)v;
    }
}

// ---------------------------------------------------------------------------
// K2/K4: conv1x1 GEMM.  Y[b,co,n] = bias[co] + sum_ci W[co,ci] * X[b,ci,n]
// Block = 256 threads (8 waves) owning a 64(co) x 128(n) tile.
// Per 32-K step: activations staged transposed in LDS (vector global loads +
// packed b32 LDS stores), then each wave does 4 WMMAs sharing one A fragment.
// RESID: f32 output + residual; else f16 output.
// ---------------------------------------------------------------------------
template <bool RESID>
__global__ __launch_bounds__(256) void gemm_kernel(
    const _Float16* __restrict__ W, const float* __restrict__ bias,
    const _Float16* __restrict__ X, void* __restrict__ Yv,
    const float* __restrict__ resid) {
    __shared__ _Float16 xT[128][32];  // xT[n_local][ci_local], 8 KB

    const int tid   = threadIdx.x;
    const int lane  = tid & 31;
    const int wave  = tid >> 5;
    const int nlane = lane & 15;
    const int hi    = lane >> 4;

    const int nblk = NHW / 128;  // 32
    int b   = blockIdx.x / ((CCH / 64) * nblk);
    int rem = blockIdx.x % ((CCH / 64) * nblk);
    int co0 = (rem / nblk) * 64;
    int n0  = (rem % nblk) * 128;

    const int co_w = co0 + (wave & 3) * 16;   // this wave's 16 out-channels
    const int n_w  = (wave >> 2) * 64;        // this wave's 64-n group (local)

    const _Float16* Xb   = X + (size_t)b * CCH * NHW;
    const _Float16* Wrow = W + (size_t)(co_w + nlane) * CCH;

    // staging assignment: each thread moves a 2(ci) x 8(n) chunk per K step
    const int ci2 = (tid >> 4) * 2;   // 0..30
    const int nn0 = (tid & 15) * 8;   // 0..120

    float binit[8];
#pragma unroll
    for (int r = 0; r < 8; ++r) binit[r] = bias[co_w + r + hi * 8];
    v8f acc[4];
#pragma unroll
    for (int t = 0; t < 4; ++t)
#pragma unroll
        for (int r = 0; r < 8; ++r) acc[t][r] = binit[r];

    for (int k0 = 0; k0 < CCH; k0 += 32) {
        __syncthreads();  // previous iteration's B reads complete
        v8h g0 = *(const v8h*)(Xb + (size_t)(k0 + ci2) * NHW + n0 + nn0);
        v8h g1 = *(const v8h*)(Xb + (size_t)(k0 + ci2 + 1) * NHW + n0 + nn0);
#pragma unroll
        for (int jj = 0; jj < 8; ++jj) {
            v2h pr; pr[0] = g0[jj]; pr[1] = g1[jj];
            *(v2h*)&xT[nn0 + jj][ci2] = pr;
        }
        __syncthreads();

        v16h a = load_a_frag(Wrow + k0 + hi * 16);
#pragma unroll
        for (int t = 0; t < 4; ++t) {
            v16h bf = *(const v16h*)&xT[n_w + t * 16 + nlane][hi * 16];
            acc[t] = wmma_f16(a, bf, acc[t]);
        }
    }

    if (RESID) {
        float* Y = (float*)Yv;
#pragma unroll
        for (int t = 0; t < 4; ++t)
#pragma unroll
            for (int r = 0; r < 8; ++r) {
                size_t idx = ((size_t)b * CCH + co_w + r + hi * 8) * NHW +
                             n0 + n_w + t * 16 + nlane;
                Y[idx] = acc[t][r] + resid[idx];
            }
    } else {
        _Float16* Y = (_Float16*)Yv;
#pragma unroll
        for (int t = 0; t < 4; ++t)
#pragma unroll
            for (int r = 0; r < 8; ++r) {
                size_t idx = ((size_t)b * CCH + co_w + r + hi * 8) * NHW +
                             n0 + n_w + t * 16 + nlane;
                Y[idx] = (_Float16)acc[t][r];
            }
    }
}

// ---------------------------------------------------------------------------
// K3: flash attention.  One block per (b, 16-row i-tile).  9 waves:
//   wave 8  : S = Q^T K (WMMA), online softmax, publishes P/alpha via LDS
//   waves0-7: each owns 64 channels of V, accumulates O += V * P^T (WMMA)
// ---------------------------------------------------------------------------
__global__ __launch_bounds__(288) void attn_kernel(
    const _Float16* __restrict__ q, const _Float16* __restrict__ k,
    const _Float16* __restrict__ v, _Float16* __restrict__ o) {
    __shared__ _Float16 qT[16][CCH];    // 16 KB  qT[i][c] = q[c][i0+i]
    __shared__ _Float16 kT[JT][CCH];    // 32 KB  kT[j][c] = k[c][j0+j]
    __shared__ _Float16 pT[16][JT];     // 1 KB   pT[i][j]
    __shared__ float alpha_s[16];
    __shared__ float linv_s[16];

    const int bid = blockIdx.x;
    const int b   = bid / (NHW / 16);
    const int i0  = (bid % (NHW / 16)) * 16;
    const _Float16* qb = q + (size_t)b * CCH * NHW;
    const _Float16* kb = k + (size_t)b * CCH * NHW;
    const _Float16* vb = v + (size_t)b * CCH * NHW;

    const int tid   = threadIdx.x;
    const int lane  = tid & 31;
    const int wave  = tid >> 5;
    const int nlane = lane & 15;
    const int hi    = lane >> 4;
    const int c0    = wave * 64;  // V waves only

    // stage Q transposed once: 2(c) x 8(i) chunks, packed b32 LDS stores
    for (int cidx = tid; cidx < 512; cidx += 288) {
        int ii0 = (cidx & 1) * 8;
        int c2  = (cidx >> 1) * 2;
        v8h g0 = *(const v8h*)(qb + (size_t)c2 * NHW + i0 + ii0);
        v8h g1 = *(const v8h*)(qb + (size_t)(c2 + 1) * NHW + i0 + ii0);
#pragma unroll
        for (int jj = 0; jj < 8; ++jj) {
            v2h pr; pr[0] = g0[jj]; pr[1] = g1[jj];
            *(v2h*)&qT[ii0 + jj][c2] = pr;
        }
    }
    __syncthreads();

    const float scale = 0.044194173824159216f;  // 512^-0.5
    v8f acc[4] = {v8f{}, v8f{}, v8f{}, v8f{}};
    float mrun[8], lrun[8];
#pragma unroll
    for (int r = 0; r < 8; ++r) { mrun[r] = -3.0e38f; lrun[r] = 0.f; }

    for (int j0 = 0; j0 < NHW; j0 += JT) {
        // cooperative transposed stage of K tile: 2(c) x 8(j) chunks
        for (int cidx = tid; cidx < 1024; cidx += 288) {
            int jj0 = (cidx & 3) * 8;
            int c2  = (cidx >> 2) * 2;
            v8h g0 = *(const v8h*)(kb + (size_t)c2 * NHW + j0 + jj0);
            v8h g1 = *(const v8h*)(kb + (size_t)(c2 + 1) * NHW + j0 + jj0);
#pragma unroll
            for (int jj = 0; jj < 8; ++jj) {
                v2h pr; pr[0] = g0[jj]; pr[1] = g1[jj];
                *(v2h*)&kT[jj0 + jj][c2] = pr;
            }
        }
        __syncthreads();

        v16h va[4];
        if (wave == 8) {
            // ---- S = Q^T K  (two 16x16 halves over j) ----
            v8f s0 = {}, s1 = {};
            for (int k0 = 0; k0 < CCH; k0 += 32) {
                v16h a  = load_a_frag(&qT[nlane][k0 + hi * 16]);
                v16h b0 = *(const v16h*)&kT[nlane][k0 + hi * 16];
                v16h b1 = *(const v16h*)&kT[16 + nlane][k0 + hi * 16];
                s0 = wmma_f16(a, b0, s0);
                s1 = wmma_f16(a, b1, s1);
            }
            // ---- online softmax over 32 j columns ----
#pragma unroll
            for (int r = 0; r < 8; ++r) {
                float x0 = s0[r] * scale, x1 = s1[r] * scale;
                float mx = fmaxf(x0, x1);
#pragma unroll
                for (int m = 8; m >= 1; m >>= 1)
                    mx = fmaxf(mx, __shfl_xor(mx, m, 32));  // within 16-lane half
                float mnew = fmaxf(mrun[r], mx);
                float al   = __expf(mrun[r] - mnew);
                float p0   = __expf(x0 - mnew);
                float p1   = __expf(x1 - mnew);
                float ps   = p0 + p1;
#pragma unroll
                for (int m = 8; m >= 1; m >>= 1)
                    ps += __shfl_xor(ps, m, 32);
                lrun[r] = lrun[r] * al + ps;
                mrun[r] = mnew;
                int row = r + hi * 8;
                pT[row][nlane]      = (_Float16)p0;
                pT[row][16 + nlane] = (_Float16)p1;
                if (nlane == 0) alpha_s[row] = al;
            }
        } else {
            // V waves: prefetch V A-fragments while softmax wave computes S
#pragma unroll
            for (int t = 0; t < 4; ++t)
                va[t] = load_a_frag(vb + (size_t)(c0 + t * 16 + nlane) * NHW +
                                    j0 + hi * 16);
        }
        __syncthreads();

        if (wave < 8) {
            float al = alpha_s[nlane];
            // B = P^T: K = j = e + 16*hi, N = i = nlane  -> pT[nlane][e+16*hi]
            v16h pb = *(const v16h*)&pT[nlane][hi * 16];
#pragma unroll
            for (int t = 0; t < 4; ++t) {
#pragma unroll
                for (int r = 0; r < 8; ++r) acc[t][r] *= al;
                acc[t] = wmma_f16(va[t], pb, acc[t]);
            }
        }
        __syncthreads();  // P/alpha consumed before next iteration overwrites
    }

    if (wave == 8) {
#pragma unroll
        for (int r = 0; r < 8; ++r)
            if (nlane == 0) linv_s[r + hi * 8] = 1.0f / lrun[r];
    }
    __syncthreads();

    if (wave < 8) {
        float li = linv_s[nlane];
        _Float16* ob = o + (size_t)b * CCH * NHW;
#pragma unroll
        for (int t = 0; t < 4; ++t)
#pragma unroll
            for (int r = 0; r < 8; ++r)
                ob[(size_t)(c0 + t * 16 + r + hi * 8) * NHW + i0 + nlane] =
                    (_Float16)(acc[t][r] * li);
    }
}

// ---------------------------------------------------------------------------
extern "C" void kernel_launch(void* const* d_in, const int* in_sizes, int n_in,
                              void* d_out, int out_size, void* d_ws, size_t ws_size,
                              hipStream_t stream) {
    const float* x     = (const float*)d_in[0];
    const float* gamma = (const float*)d_in[1];
    const float* beta  = (const float*)d_in[2];
    const float* q_w   = (const float*)d_in[3];
    const float* q_b   = (const float*)d_in[4];
    const float* k_w   = (const float*)d_in[5];
    const float* k_b   = (const float*)d_in[6];
    const float* v_w   = (const float*)d_in[7];
    const float* v_b   = (const float*)d_in[8];
    const float* p_w   = (const float*)d_in[9];
    const float* p_b   = (const float*)d_in[10];
    float* out = (float*)d_out;

    const size_t ACT = (size_t)BATCH * CCH * NHW;  // 4M f16 elems per buffer
    const size_t WSZ = (size_t)CCH * CCH;          // 256K f16 elems per weight
    _Float16* hn = (_Float16*)d_ws;
    _Float16* qf = hn + ACT;
    _Float16* kf = qf + ACT;
    _Float16* vf = kf + ACT;
    _Float16* of = vf + ACT;
    _Float16* wq = of + ACT;
    _Float16* wk = wq + WSZ;
    _Float16* wv = wk + WSZ;
    _Float16* wp = wv + WSZ;

    // K0: weight conversion
    {
        int n = (int)WSZ, g = (n + 255) / 256;
        cvt_f16_kernel<<<g, 256, 0, stream>>>(q_w, wq, n);
        cvt_f16_kernel<<<g, 256, 0, stream>>>(k_w, wk, n);
        cvt_f16_kernel<<<g, 256, 0, stream>>>(v_w, wv, n);
        cvt_f16_kernel<<<g, 256, 0, stream>>>(p_w, wp, n);
    }
    // K1: group norm
    groupnorm_kernel<<<BATCH * NGRP, 256, 0, stream>>>(x, gamma, beta, hn);

    // K2: q, k, v projections (64x128 tile per block)
    const int gemm_blocks = BATCH * (CCH / 64) * (NHW / 128);  // 512
    gemm_kernel<false><<<gemm_blocks, 256, 0, stream>>>(wq, q_b, hn, (void*)qf, nullptr);
    gemm_kernel<false><<<gemm_blocks, 256, 0, stream>>>(wk, k_b, hn, (void*)kf, nullptr);
    gemm_kernel<false><<<gemm_blocks, 256, 0, stream>>>(wv, v_b, hn, (void*)vf, nullptr);

    // K3: flash attention
    attn_kernel<<<BATCH * (NHW / 16), 288, 0, stream>>>(qf, kf, vf, of);

    // K4: output projection + residual (f32 out)
    gemm_kernel<true><<<gemm_blocks, 256, 0, stream>>>(wp, p_b, of, (void*)out, x);
}
